// FastGNNLinearPrecoding_48060684042857
// MI455X (gfx1250) — matrix operand order repaired and checked by reading, not compiled, not run on hardware
//
#include <hip/hip_runtime.h>
#include <hip/hip_bf16.h>

#define NN 50000
#define NE 800000
#define DD 128

typedef __attribute__((ext_vector_type(16))) __bf16 v16bf;
typedef __attribute__((ext_vector_type(8)))  float  v8f;
typedef __attribute__((ext_vector_type(8)))  unsigned short u16x8;

// ---------------- f32 -> bf16 (round-nearest-even) ----------------
__global__ void cvt_bf16_kernel(const float* __restrict__ in,
                                unsigned short* __restrict__ out, int n) {
    int i = blockIdx.x * blockDim.x + threadIdx.x;
    if (i < n) {
        unsigned u = __builtin_bit_cast(unsigned, in[i]);
        unsigned r = u + 0x7FFFu + ((u >> 16) & 1u);
        out[i] = (unsigned short)(r >> 16);
    }
}

// ---------------- 8-way projection GEMM via WMMA bf16 ----------------
// xs[k][n][o] = sum_d x[n][d] * W[k][o][d] + bias[k][o]
// One wave per 16x16 output tile; block = 8 waves = 8 o-tiles of same (k, n_tile).
__global__ void proj_wmma_kernel(const unsigned short* __restrict__ xbf,  // [NN,128] bf16
                                 const unsigned short* __restrict__ wbf,  // [8,128,128] bf16
                                 const float* __restrict__ bias,          // [8,128]
                                 float* __restrict__ xs)                  // [8,NN,128]
{
    const int lane = threadIdx.x & 31;
    const int wave = threadIdx.x >> 5;           // o-tile 0..7
    const int k      = blockIdx.x / (NN / 16);   // weight index 0..7
    const int n_tile = blockIdx.x % (NN / 16);
    const int nb = n_tile * 16;
    const int ob = wave * 16;
    const int lrow = lane & 15;
    const int half = lane >> 4;

    const unsigned short* arow = xbf + (size_t)(nb + lrow) * DD;
    const unsigned short* brow = wbf + ((size_t)k * DD + (ob + lrow)) * DD;

    v8f c = {};
    #pragma unroll
    for (int d = 0; d < DD; d += 32) {
        // A fragment (16x32 bf16): lane holds K = {half*8..+7} and {16+half*8..+7}
        u16x8 a0 = *(const u16x8*)(arow + d + half * 8);
        u16x8 a1 = *(const u16x8*)(arow + d + 16 + half * 8);
        // B fragment (32x16 bf16): lane holds K = {half*8..+7} and {16+half*8..+7}
        u16x8 b0 = *(const u16x8*)(brow + d + half * 8);
        u16x8 b1 = *(const u16x8*)(brow + d + 16 + half * 8);
        v16bf A, B;
        #pragma unroll
        for (int i = 0; i < 8; ++i) {
            A[i]     = __builtin_bit_cast(__bf16, (unsigned short)a0[i]);
            A[i + 8] = __builtin_bit_cast(__bf16, (unsigned short)a1[i]);
            B[i]     = __builtin_bit_cast(__bf16, (unsigned short)b0[i]);
            B[i + 8] = __builtin_bit_cast(__bf16, (unsigned short)b1[i]);
        }
        c = __builtin_amdgcn_wmma_f32_16x16x32_bf16(
                /*neg_a=*/false, A, /*neg_b=*/false, B,
                /*c_mod=*/(short)0, c, /*reuse_a=*/false, /*reuse_b=*/false);
    }
    const float bv = bias[k * DD + ob + lrow];
    #pragma unroll
    for (int r = 0; r < 8; ++r) {
        const int row = nb + r + 8 * half;               // C/D layout: M = r + 8*(lane/16)
        xs[((size_t)k * NN + row) * DD + ob + lrow] = c[r] + bv;
    }
}

// ---------------- fused edge pass: unnormalized softmax-aggregate ----------------
// agg[i] += exp(score_e) * x_val[j];  den[i] += exp(score_e)
__global__ void edge_attn_kernel(const float* __restrict__ xval,
                                 const float* __restrict__ xdst,
                                 const float* __restrict__ xsrc,
                                 const int*   __restrict__ edges,   // [2,NE]: row 0 = j(src), row 1 = i(dst)
                                 float* __restrict__ agg,           // [NN,128]
                                 float* __restrict__ den)           // [NN]
{
    const int lane = threadIdx.x & 31;
    const int e = (blockIdx.x * blockDim.x + threadIdx.x) >> 5;
    if (e >= NE) return;
    const int j = edges[e];
    const int i = edges[NE + e];

    const float4 d4 = *(const float4*)(xdst + (size_t)i * DD + lane * 4);
    const float4 s4 = *(const float4*)(xsrc + (size_t)j * DD + lane * 4);
    float p = d4.x * s4.x + d4.y * s4.y + d4.z * s4.z + d4.w * s4.w;
    #pragma unroll
    for (int o = 16; o > 0; o >>= 1) p += __shfl_xor(p, o, 32);

    const float num = __expf(p * 0.08838834764831845f);   // / sqrt(128)
    if (lane == 0) unsafeAtomicAdd(den + i, num);

    const float4 v4 = *(const float4*)(xval + (size_t)j * DD + lane * 4);
    float* a = agg + (size_t)i * DD + lane * 4;
    unsafeAtomicAdd(a + 0, num * v4.x);
    unsafeAtomicAdd(a + 1, num * v4.y);
    unsafeAtomicAdd(a + 2, num * v4.z);
    unsafeAtomicAdd(a + 3, num * v4.w);
}

// ---------------- combine + ReLU + LayerNorm (wave per node) ----------------
__global__ void combine_ln_kernel(const float* __restrict__ xs,
                                  const float* __restrict__ agg_ue, const float* __restrict__ den_ue,
                                  const float* __restrict__ agg_ap, const float* __restrict__ den_ap,
                                  const float* __restrict__ g, const float* __restrict__ b,
                                  float* __restrict__ xout)
{
    const int lane = threadIdx.x & 31;
    const int n = (blockIdx.x * blockDim.x + threadIdx.x) >> 5;
    if (n >= NN) return;
    const size_t off = (size_t)n * DD + lane * 4;

    const float4 x1 = *(const float4*)(xs + off);                                // k = 0
    const float4 x5 = *(const float4*)(xs + (size_t)4 * NN * DD + off);          // k = 4
    const float4 au = *(const float4*)(agg_ue + off);
    const float4 aa = *(const float4*)(agg_ap + off);
    const float du = den_ue[n], da = den_ap[n];
    const float iu = du > 0.f ? 1.f / du : 0.f;
    const float ia = da > 0.f ? 1.f / da : 0.f;

    float h[4];
    h[0] = fmaxf(au.x * iu + x1.x + aa.x * ia + x5.x, 0.f);
    h[1] = fmaxf(au.y * iu + x1.y + aa.y * ia + x5.y, 0.f);
    h[2] = fmaxf(au.z * iu + x1.z + aa.z * ia + x5.z, 0.f);
    h[3] = fmaxf(au.w * iu + x1.w + aa.w * ia + x5.w, 0.f);

    float s1 = h[0] + h[1] + h[2] + h[3];
    float s2 = h[0]*h[0] + h[1]*h[1] + h[2]*h[2] + h[3]*h[3];
    #pragma unroll
    for (int o = 16; o > 0; o >>= 1) {
        s1 += __shfl_xor(s1, o, 32);
        s2 += __shfl_xor(s2, o, 32);
    }
    const float mu  = s1 * (1.0f / DD);
    const float var = s2 * (1.0f / DD) - mu * mu;
    const float rs  = rsqrtf(var + 1e-5f);

    float4 outv;
    const int f = lane * 4;
    outv.x = (h[0] - mu) * rs * g[f + 0] + b[f + 0];
    outv.y = (h[1] - mu) * rs * g[f + 1] + b[f + 1];
    outv.z = (h[2] - mu) * rs * g[f + 2] + b[f + 2];
    outv.w = (h[3] - mu) * rs * g[f + 3] + b[f + 3];
    *(float4*)(xout + off) = outv;
}

// ---------------- final linear [128 -> 6] (wave per node) ----------------
__global__ void final_lin_kernel(const float* __restrict__ x,
                                 const float* __restrict__ w,   // [6,128]
                                 const float* __restrict__ bias,// [6]
                                 float* __restrict__ out)       // [NN,6]
{
    const int lane = threadIdx.x & 31;
    const int n = (blockIdx.x * blockDim.x + threadIdx.x) >> 5;
    if (n >= NN) return;
    const float4 xv = *(const float4*)(x + (size_t)n * DD + lane * 4);

    float acc[6];
    #pragma unroll
    for (int c = 0; c < 6; ++c) {
        const float4 wv = *(const float4*)(w + c * DD + lane * 4);
        float p = xv.x * wv.x + xv.y * wv.y + xv.z * wv.z + xv.w * wv.w;
        #pragma unroll
        for (int o = 16; o > 0; o >>= 1) p += __shfl_xor(p, o, 32);
        acc[c] = p;
    }
    if (lane == 0) {
        #pragma unroll
        for (int c = 0; c < 6; ++c) out[(size_t)n * 6 + c] = acc[c] + bias[c];
    }
}

extern "C" void kernel_launch(void* const* d_in, const int* in_sizes, int n_in,
                              void* d_out, int out_size, void* d_ws, size_t ws_size,
                              hipStream_t stream) {
    const float* x0    = (const float*)d_in[0];
    const int*   e_ue  = (const int*)  d_in[1];
    const int*   e_ap  = (const int*)  d_in[2];
    const float* Ws    = (const float*)d_in[3];   // [2,8,128,128]
    const float* bs    = (const float*)d_in[4];   // [2,8,128]
    const float* ln_g  = (const float*)d_in[5];   // [2,128]
    const float* ln_b  = (const float*)d_in[6];   // [2,128]
    const float* lin_w = (const float*)d_in[7];   // [6,128]
    const float* lin_b = (const float*)d_in[8];   // [6]
    float* out = (float*)d_out;

    char* ws = (char*)d_ws;
    const size_t ND = (size_t)NN * DD;            // 6.4M elems
    float*          xs     = (float*)(ws);                                   // 8*ND f32
    float*          xA     = (float*)(ws + 8 * ND * 4);                      // ND f32
    float*          xB     = (float*)(ws + 9 * ND * 4);                      // ND f32
    float*          agg_ue = (float*)(ws + 10 * ND * 4);                     // ND f32
    float*          agg_ap = (float*)(ws + 11 * ND * 4);                     // ND f32
    float*          den_ue = (float*)(ws + 12 * ND * 4);                     // NN f32
    float*          den_ap = (float*)(ws + 12 * ND * 4 + NN * 4);            // NN f32
    unsigned short* xbf    = (unsigned short*)(ws + 12 * ND * 4 + 2 * (size_t)NN * 4);
    unsigned short* wbf    = (unsigned short*)((char*)xbf + ND * 2);         // 8*128*128 bf16

    const float* xin = x0;
    float* xout_buf[2] = { xA, xB };

    for (int l = 0; l < 2; ++l) {
        cvt_bf16_kernel<<<(int)((ND + 255) / 256), 256, 0, stream>>>(xin, xbf, (int)ND);
        cvt_bf16_kernel<<<(8 * DD * DD + 255) / 256, 256, 0, stream>>>(
            Ws + (size_t)l * 8 * DD * DD, wbf, 8 * DD * DD);

        hipMemsetAsync(agg_ue, 0, ND * 4, stream);
        hipMemsetAsync(agg_ap, 0, ND * 4, stream);
        hipMemsetAsync(den_ue, 0, (size_t)NN * 4, stream);
        hipMemsetAsync(den_ap, 0, (size_t)NN * 4, stream);

        proj_wmma_kernel<<<8 * (NN / 16), 256, 0, stream>>>(
            xbf, wbf, bs + (size_t)l * 8 * DD, xs);

        // ue graph: val=x2(k=1), dst=x3(k=2), src=x4(k=3)
        edge_attn_kernel<<<NE / 8, 256, 0, stream>>>(
            xs + 1 * ND, xs + 2 * ND, xs + 3 * ND, e_ue, agg_ue, den_ue);
        // ap graph: val=x6(k=5), dst=x7(k=6), src=x8(k=7)
        edge_attn_kernel<<<NE / 8, 256, 0, stream>>>(
            xs + 5 * ND, xs + 6 * ND, xs + 7 * ND, e_ap, agg_ap, den_ap);

        combine_ln_kernel<<<(NN * 32) / 256, 256, 0, stream>>>(
            xs, agg_ue, den_ue, agg_ap, den_ap,
            ln_g + l * DD, ln_b + l * DD, xout_buf[l]);

        xin = xout_buf[l];
    }

    final_lin_kernel<<<(NN * 32) / 256, 256, 0, stream>>>(xin, lin_w, lin_b, out);
}
